// MultiHeadAttention_65996467470971
// MI455X (gfx1250) — matrix operand order repaired and checked by reading, not compiled, not run on hardware
//
#include <hip/hip_runtime.h>
#include <hip/hip_bf16.h>

// MI455X (gfx1250) MHA forward. All matmuls via v_wmma_f32_16x16x32_bf16.
// DIM=1024, H=16, hd=64, B=2, N=2048, S=B*N=4096. SCALE=0.125 folded into Q.
//
// Workspace layout (bf16 as ushort), total 48 MB (assumed <= ws_size):
//   [ 0MB) xb   : x bf16            [4096][1024]
//   [ 8MB) wqt  : Wqkv^T bf16       [3072][1024]
//   [14MB) wpt  : Wproj^T bf16      [1024][1024]
//   [16MB) qb   : Q*scale bf16      [B,H,N,hd] = [32][2048][64]
//   [24MB) kb   : K bf16            [32][2048][64]
//   [32MB) vtb  : V^T bf16          [32][64][2048]
//   [40MB) ctx  : attn output bf16  [4096][1024]

typedef __attribute__((ext_vector_type(16))) __bf16 v16bf;
typedef __attribute__((ext_vector_type(8)))  float  v8f;

union Frag16 {
    v16bf v;
    uint4 q[2];
};

__device__ __forceinline__ unsigned short f2bf(float f) {
    unsigned int u = __float_as_uint(f);
    u += 0x7FFFu + ((u >> 16) & 1u);          // round-to-nearest-even
    return (unsigned short)(u >> 16);
}

__device__ __forceinline__ uint4 ld16(const unsigned short* p) {
    return *reinterpret_cast<const uint4*>(p);
}

__device__ __forceinline__ v8f vzero() {
    v8f z;
#pragma unroll
    for (int i = 0; i < 8; ++i) z[i] = 0.0f;
    return z;
}

__device__ __forceinline__ v8f wmma_bf16(v16bf a, v16bf b, v8f c) {
    return __builtin_amdgcn_wmma_f32_16x16x32_bf16(false, a, false, b,
                                                   (short)0, c, false, false);
}

// ---------------- precision conversion / transpose ----------------

__global__ void __launch_bounds__(256) k_cvt(const float* __restrict__ in,
                                             unsigned short* __restrict__ out, int n) {
    int i = blockIdx.x * 256 + threadIdx.x;
    if (i < n) out[i] = f2bf(in[i]);
}

// in[K][N] fp32 -> out[N][K] bf16 (output-coalesced; reads hit L2)
__global__ void __launch_bounds__(256) k_cvt_t(const float* __restrict__ in,
                                               unsigned short* __restrict__ out,
                                               int K, int N) {
    int i = blockIdx.x * 256 + threadIdx.x;
    if (i < K * N) {
        int n = i / K, k = i % K;
        out[i] = f2bf(in[k * N + n]);
    }
}

// ---------------- GEMM1: qkv = xb @ Wqkv, split-store Q*scale / K / V^T ---------

__global__ void __launch_bounds__(256) k_qkv_gemm(const unsigned short* __restrict__ xb,
                                                  const unsigned short* __restrict__ wt,
                                                  unsigned short* __restrict__ qb,
                                                  unsigned short* __restrict__ kb,
                                                  unsigned short* __restrict__ vtb) {
    const int lane = threadIdx.x & 31, wave = threadIdx.x >> 5;
    const int half = lane >> 4, lr = lane & 15;
    const int row0 = blockIdx.y * 64 + (wave >> 1) * 16;   // wave: 16 rows
    const int col0 = blockIdx.x * 128 + (wave & 1) * 64;   // wave: 64 cols (4 n-tiles)

    v8f acc[4];
#pragma unroll
    for (int t = 0; t < 4; ++t) acc[t] = vzero();

    const unsigned short* arow = xb + (row0 + lr) * 1024;
    for (int k0 = 0; k0 < 1024; k0 += 32) {
        Frag16 a;
        a.q[0] = ld16(arow + k0 + 16 * half);        // A elems 0..7 : k = e + 16h
        a.q[1] = ld16(arow + k0 + 16 * half + 8);    // A elems 8..15: k = e + 16h
#pragma unroll
        for (int t = 0; t < 4; ++t) {
            const unsigned short* bp = wt + (col0 + t * 16 + lr) * 1024 + k0 + 16 * half;
            Frag16 b;
            b.q[0] = ld16(bp);                       // B elems e: k = e + 16h
            b.q[1] = ld16(bp + 8);
            acc[t] = wmma_bf16(a.v, b.v, acc[t]);
        }
    }

#pragma unroll
    for (int t = 0; t < 4; ++t) {
#pragma unroll
        for (int r = 0; r < 8; ++r) {
            int srow = row0 + r + 8 * half;          // C layout: M = r + 8*half
            int scol = col0 + t * 16 + lr;           //           N = lane&15
            int b_ = srow >> 11, n = srow & 2047;
            int which = scol >> 10, rem = scol & 1023;
            int h = rem >> 6, d = rem & 63;
            float val = acc[t][r];
            if (which == 0) {
                qb[((b_ * 16 + h) * 2048 + n) * 64 + d] = f2bf(val * 0.125f); // fold scale
            } else if (which == 1) {
                kb[((b_ * 16 + h) * 2048 + n) * 64 + d] = f2bf(val);
            } else {
                vtb[((b_ * 16 + h) * 64 + d) * 2048 + n] = f2bf(val);         // V^T
            }
        }
    }
}

// ---------------- flash attention: per (b,h), 128 queries / block ----------------

__global__ void __launch_bounds__(256) k_attn(const unsigned short* __restrict__ qb,
                                              const unsigned short* __restrict__ kb,
                                              const unsigned short* __restrict__ vtb,
                                              unsigned short* __restrict__ ctx) {
    // per-wave P tile (16 rows x 32 keys), padded to 40 to dodge bank conflicts
    __shared__ __align__(16) unsigned short pshm[8][16][40];

    const int lane = threadIdx.x & 31, wave = threadIdx.x >> 5;
    const int half = lane >> 4, lr = lane & 15;
    const int bh = blockIdx.y;                 // 0..31
    const int b_ = bh >> 4, h = bh & 15;
    const int q0 = blockIdx.x * 128 + wave * 16;

    const unsigned short* qp = qb + (bh * 2048 + q0 + lr) * 64;
    const unsigned short* kp = kb + bh * 2048 * 64;
    const unsigned short* vp = vtb + bh * 64 * 2048;

    // Q fragments (hd=64 -> two K=32 chunks), resident for whole kernel
    Frag16 aq0, aq1;
    aq0.q[0] = ld16(qp + 16 * half);
    aq0.q[1] = ld16(qp + 8 + 16 * half);
    aq1.q[0] = ld16(qp + 32 + 16 * half);
    aq1.q[1] = ld16(qp + 40 + 16 * half);

    float m[8], l[8];
    v8f o[4];
#pragma unroll
    for (int r = 0; r < 8; ++r) { m[r] = -3.0e38f; l[r] = 0.0f; }
#pragma unroll
    for (int f = 0; f < 4; ++f) o[f] = vzero();

    for (int j0 = 0; j0 < 2048; j0 += 32) {
        // scores for 32 keys = two 16-key N-tiles, each 2 WMMAs over hd chunks
        Frag16 bk;
        v8f s0 = vzero(), s1 = vzero();
        const unsigned short* kr0 = kp + (j0 + lr) * 64 + 16 * half;
        bk.q[0] = ld16(kr0);        bk.q[1] = ld16(kr0 + 8);
        s0 = wmma_bf16(aq0.v, bk.v, s0);
        bk.q[0] = ld16(kr0 + 32);   bk.q[1] = ld16(kr0 + 40);
        s0 = wmma_bf16(aq1.v, bk.v, s0);
        const unsigned short* kr1 = kp + (j0 + 16 + lr) * 64 + 16 * half;
        bk.q[0] = ld16(kr1);        bk.q[1] = ld16(kr1 + 8);
        s1 = wmma_bf16(aq0.v, bk.v, s1);
        bk.q[0] = ld16(kr1 + 32);   bk.q[1] = ld16(kr1 + 40);
        s1 = wmma_bf16(aq1.v, bk.v, s1);

        // online softmax; row M = r + 8*half lives across the 16-lane half-group
#pragma unroll
        for (int r = 0; r < 8; ++r) {
            float mx = fmaxf(s0[r], s1[r]);
            mx = fmaxf(mx, __shfl_xor(mx, 1));
            mx = fmaxf(mx, __shfl_xor(mx, 2));
            mx = fmaxf(mx, __shfl_xor(mx, 4));
            mx = fmaxf(mx, __shfl_xor(mx, 8));
            float nm = fmaxf(m[r], mx);
            float p0 = __expf(s0[r] - nm);
            float p1 = __expf(s1[r] - nm);
            float sum = p0 + p1;
            sum += __shfl_xor(sum, 1);
            sum += __shfl_xor(sum, 2);
            sum += __shfl_xor(sum, 4);
            sum += __shfl_xor(sum, 8);
            float al = __expf(m[r] - nm);
            m[r] = nm;
            l[r] = l[r] * al + sum;
            o[0][r] *= al; o[1][r] *= al; o[2][r] *= al; o[3][r] *= al;
            // C-layout -> LDS [row][key]
            pshm[wave][r + 8 * half][lr] = f2bf(p0);
            pshm[wave][r + 8 * half][16 + lr] = f2bf(p1);
        }
        __syncthreads();

        // re-read P in A-fragment layout (lane=row)
        Frag16 pa;
        pa.q[0] = ld16(&pshm[wave][lr][16 * half]);
        pa.q[1] = ld16(&pshm[wave][lr][16 * half + 8]);

        // O += P(16x32) @ V(32x64) : 4 d-tiles, V^T gives contiguous B frags
#pragma unroll
        for (int f = 0; f < 4; ++f) {
            const unsigned short* vr = vp + (f * 16 + lr) * 2048 + j0 + 16 * half;
            Frag16 bv;
            bv.q[0] = ld16(vr);
            bv.q[1] = ld16(vr + 8);
            o[f] = wmma_bf16(pa.v, bv.v, o[f]);
        }
        __syncthreads();   // WAR on pshm before next iteration
    }

    // normalize and store ctx as [S][1024] bf16 (A operand of the proj GEMM)
#pragma unroll
    for (int f = 0; f < 4; ++f) {
#pragma unroll
        for (int r = 0; r < 8; ++r) {
            int n = q0 + r + 8 * half;
            int d = f * 16 + lr;
            ctx[(b_ * 2048 + n) * 1024 + h * 64 + d] = f2bf(o[f][r] / l[r]);
        }
    }
}

// ---------------- GEMM2: out = ctx @ Wproj + b (fp32 out) ----------------

__global__ void __launch_bounds__(256) k_proj_gemm(const unsigned short* __restrict__ ctx,
                                                   const unsigned short* __restrict__ wpt,
                                                   const float* __restrict__ bias,
                                                   float* __restrict__ out) {
    const int lane = threadIdx.x & 31, wave = threadIdx.x >> 5;
    const int half = lane >> 4, lr = lane & 15;
    const int row0 = blockIdx.y * 64 + (wave >> 1) * 16;
    const int col0 = blockIdx.x * 128 + (wave & 1) * 64;

    v8f acc[4];
#pragma unroll
    for (int t = 0; t < 4; ++t) acc[t] = vzero();

    const unsigned short* arow = ctx + (row0 + lr) * 1024;
    for (int k0 = 0; k0 < 1024; k0 += 32) {
        Frag16 a;
        a.q[0] = ld16(arow + k0 + 16 * half);
        a.q[1] = ld16(arow + k0 + 16 * half + 8);
#pragma unroll
        for (int t = 0; t < 4; ++t) {
            const unsigned short* bp = wpt + (col0 + t * 16 + lr) * 1024 + k0 + 16 * half;
            Frag16 b;
            b.q[0] = ld16(bp);
            b.q[1] = ld16(bp + 8);
            acc[t] = wmma_bf16(a.v, b.v, acc[t]);
        }
    }

#pragma unroll
    for (int t = 0; t < 4; ++t) {
#pragma unroll
        for (int r = 0; r < 8; ++r) {
            int srow = row0 + r + 8 * half;
            int scol = col0 + t * 16 + lr;
            out[srow * 1024 + scol] = acc[t][r] + bias[scol];
        }
    }
}

// ---------------- host launcher ----------------

extern "C" void kernel_launch(void* const* d_in, const int* in_sizes, int n_in,
                              void* d_out, int out_size, void* d_ws, size_t ws_size,
                              hipStream_t stream) {
    const float* x     = (const float*)d_in[0];   // [2,2048,1024]
    const float* wqkv  = (const float*)d_in[1];   // [1024,3072]
    const float* wproj = (const float*)d_in[2];   // [1024,1024]
    const float* bproj = (const float*)d_in[3];   // [1024]
    float* out = (float*)d_out;

    char* ws = (char*)d_ws;
    unsigned short* xb   = (unsigned short*)(ws);
    unsigned short* wqt  = (unsigned short*)(ws + (size_t)( 8u << 20));
    unsigned short* wpt  = (unsigned short*)(ws + (size_t)(14u << 20));
    unsigned short* qb   = (unsigned short*)(ws + (size_t)(16u << 20));
    unsigned short* kb   = (unsigned short*)(ws + (size_t)(24u << 20));
    unsigned short* vtb  = (unsigned short*)(ws + (size_t)(32u << 20));
    unsigned short* ctx  = (unsigned short*)(ws + (size_t)(40u << 20));

    k_cvt  <<<(4194304 + 255) / 256, 256, 0, stream>>>(x, xb, 4194304);
    k_cvt_t<<<(3145728 + 255) / 256, 256, 0, stream>>>(wqkv, wqt, 1024, 3072);
    k_cvt_t<<<(1048576 + 255) / 256, 256, 0, stream>>>(wproj, wpt, 1024, 1024);

    k_qkv_gemm <<<dim3(24, 64), 256, 0, stream>>>(xb, wqt, qb, kb, vtb);
    k_attn     <<<dim3(16, 32), 256, 0, stream>>>(qb, kb, vtb, ctx);
    k_proj_gemm<<<dim3( 8, 64), 256, 0, stream>>>(ctx, wpt, bproj, out);
}
